// EMAStream_7713761264325
// MI455X (gfx1250) — matrix operand order repaired and checked by reading, not compiled, not run on hardware
//
#include <hip/hip_runtime.h>
#include <hip/hip_bf16.h>

// ---------------------------------------------------------------------------
// EMA (chunked parallel scan over S) + Linear (bf16 WMMA GEMM, fp32 accum).
//   x:[B,S,H] f32, W:[O,H] f32, b:[O] f32  ->  out:[B,S,O] f32
//   B=8, S=2048, H=O=2048  ->  GEMM M=16384, N=2048, K=2048
// GEMM staging uses the CDNA5 Tensor Data Mover (tensor_load_to_lds) with
// descriptor-driven LDS padding; compute is V_WMMA_F32_16X16X32_BF16.
// ---------------------------------------------------------------------------

typedef __attribute__((ext_vector_type(16))) __bf16        v16bf;
typedef __attribute__((ext_vector_type(8)))  float         v8f;
typedef __attribute__((ext_vector_type(4)))  unsigned int  u32x4;
typedef __attribute__((ext_vector_type(8)))  int           i32x8;
typedef __attribute__((ext_vector_type(4)))  int           i32x4;

#define ALPHA   0.01f
#define B_DIM   8
#define S_LEN   2048
#define H_DIM   2048
#define M_DIM   (B_DIM * S_LEN)   // 16384
#define K_DIM   H_DIM             // 2048
#define N_DIM   H_DIM             // 2048

// scan chunking: S = CHUNKS * CHUNK_LEN
#define CHUNKS     16
#define CHUNK_LEN  (S_LEN / CHUNKS)   // 128 = 2^7

// K staged per round and LDS row stride (in bf16 elems): 64 data + 8 pad.
// Row stride = 144 B (16B aligned; 36-bank stride -> conflict-free for the
// 16-consecutive-row fragment reads). TDM pad: 4 DWORDs after every 32 DWORDs.
#define K_STEP      64
#define LDS_STRIDE  72
#define A_TILE_ELEMS (128 * LDS_STRIDE)   //  9216 bf16 = 18 KiB
#define B_TILE_ELEMS (256 * LDS_STRIDE)   // 18432 bf16 = 36 KiB

// ---------------------------------------------------------------------------
// Scan pass A: per (b,chunk,h), carry-free chunk reduction
//   e starts at x_0 (chunk 0, since s_0 = x_0) or a*x_{cL} (other chunks),
//   e <- d*e + a*x ; E[b,c,h] = e at chunk end.
// 262144 threads, coalesced across h.
// ---------------------------------------------------------------------------
__global__ void ema_partial_kernel(const float* __restrict__ x,
                                   float* __restrict__ E) {
    const int idx = blockIdx.x * blockDim.x + threadIdx.x;   // 0 .. B*CHUNKS*H-1
    const int h  = idx & (H_DIM - 1);
    const int bc = idx >> 11;                 // / H_DIM
    const int c  = bc & (CHUNKS - 1);
    const int b  = bc >> 4;                   // / CHUNKS

    const float* __restrict__ xp =
        x + ((size_t)b * S_LEN + (size_t)c * CHUNK_LEN) * H_DIM + h;

    const float om = 1.0f - ALPHA;
    float e = (c == 0) ? xp[0] : (ALPHA * xp[0]);
#pragma unroll 4
    for (int t = 1; t < CHUNK_LEN; ++t)
        e = fmaf(om, e, ALPHA * xp[(size_t)t * H_DIM]);

    E[((size_t)b * CHUNKS + c) * H_DIM + h] = e;
}

// ---------------------------------------------------------------------------
// Scan pass B: carry-in S_{c-1} = fold(E_0..E_{c-1}) with d^CHUNK_LEN,
// then replay the local recurrence and emit bf16 s.
// ---------------------------------------------------------------------------
__global__ void ema_apply_kernel(const float* __restrict__ x,
                                 const float* __restrict__ E,
                                 __hip_bfloat16* __restrict__ s) {
    const int idx = blockIdx.x * blockDim.x + threadIdx.x;
    const int h  = idx & (H_DIM - 1);
    const int bc = idx >> 11;
    const int c  = bc & (CHUNKS - 1);
    const int b  = bc >> 4;

    const float om = 1.0f - ALPHA;
    // dL = om^CHUNK_LEN, CHUNK_LEN = 2^7 -> 7 squarings (exact)
    float dL = om;
#pragma unroll
    for (int i = 0; i < 7; ++i) dL *= dL;

    // carry = S_{c-1}:  S_k = dL*S_{k-1} + E_k
    float carry = 0.0f;
    for (int cp = 0; cp < c; ++cp)
        carry = fmaf(carry, dL, E[((size_t)b * CHUNKS + cp) * H_DIM + h]);

    const size_t base = ((size_t)b * S_LEN + (size_t)c * CHUNK_LEN) * H_DIM + h;
    const float* __restrict__ xp = x + base;
    __hip_bfloat16* __restrict__ sp = s + base;

    float prev = (c == 0) ? xp[0] : fmaf(om, carry, ALPHA * xp[0]);
    sp[0] = __float2bfloat16(prev);
#pragma unroll 4
    for (int t = 1; t < CHUNK_LEN; ++t) {
        prev = fmaf(om, prev, ALPHA * xp[(size_t)t * H_DIM]);
        sp[(size_t)t * H_DIM] = __float2bfloat16(prev);
    }
}

// ---------------------------------------------------------------------------
// W fp32 -> bf16.
// ---------------------------------------------------------------------------
__global__ void conv_w_kernel(const float* __restrict__ W,
                              __hip_bfloat16* __restrict__ Wb, int n) {
    const int i = blockIdx.x * blockDim.x + threadIdx.x;
    if (i < n) Wb[i] = __float2bfloat16(W[i]);
}

// ---------------------------------------------------------------------------
// TDM tile load: rows x 64 bf16 tile from a row-major [tensor_rows][2048]
// bf16 tensor into LDS at lds_off, padding 16 B after each 128-B row so the
// LDS row stride is 144 B. D# layout per CDNA5 ISA sec. 8 (groups 0/1).
// ---------------------------------------------------------------------------
__device__ __forceinline__ void tdm_load_tile(const __hip_bfloat16* gptr,
                                              unsigned lds_off,
                                              unsigned tile_rows,
                                              unsigned tensor_rows) {
    const unsigned long long ga = (unsigned long long)(uintptr_t)gptr;
    // group0: [1:0]=count=1 | [63:32]=lds_addr | [120:64]=global_addr | [127:126]=type 2
    u32x4 g0 = { 1u,
                 lds_off,
                 (unsigned)(ga & 0xFFFFFFFFu),
                 (unsigned)((ga >> 32) & 0x01FFFFFFu) | (2u << 30) };
    // group1:
    //  w0: mask=0 | data_size=1 (2B) | pad_enable | pad_interval=4 (32 dw) | pad_amount=3 (4 dw)
    //  w1[63:48] = tensor_dim0 lo16 (2048)
    //  w2[31:16] = tensor_dim1 lo16 ; w3[15:0] = tensor_dim1 hi16
    //  w3[31:16] = tile_dim0 (64)
    //  w4[15:0]  = tile_dim1 (rows)
    //  w5        = tensor_dim0_stride lo32 (2048 elems)
    i32x8 g1 = { (int)((1u << 16) | (1u << 20) | (4u << 22) | (3u << 25)),
                 (int)(2048u << 16),
                 (int)((tensor_rows & 0xFFFFu) << 16),
                 (int)(((tensor_rows >> 16) & 0xFFFFu) | (64u << 16)),
                 (int)tile_rows,
                 2048,
                 0,
                 0 };
    i32x4 gz4 = { 0, 0, 0, 0 };
    i32x8 gz8 = { 0, 0, 0, 0, 0, 0, 0, 0 };
    __builtin_amdgcn_tensor_load_to_lds(g0, g1, gz4, gz4, gz8, 0);
}

// ---------------------------------------------------------------------------
// GEMM: out[M][N] = A[M][K](bf16) * Wb[N][K](bf16)^T + bias, f32 out.
// Block: 256 threads (8 wave32s), 128(M) x 256(N) tile, wave grid 2x4,
// each wave a 64x64 subtile = 4x4 WMMA accumulators.
// K staged 64 wide by the TDM into double-buffered padded LDS.
// ---------------------------------------------------------------------------
__global__ __launch_bounds__(256)
void wmma_gemm_kernel(const __hip_bfloat16* __restrict__ A,   // [M][K]
                      const __hip_bfloat16* __restrict__ Bm,  // [N][K]
                      const float* __restrict__ bias,         // [N]
                      float* __restrict__ C) {                // [M][N]
    __shared__ __hip_bfloat16 ldsA[2 * A_TILE_ELEMS];   // 36 KiB
    __shared__ __hip_bfloat16 ldsB[2 * B_TILE_ELEMS];   // 72 KiB

    const int m_base = blockIdx.y * 128;
    const int n_base = blockIdx.x * 256;
    const int tid    = threadIdx.x;
    const int lane   = tid & 31;
    const int wid    = tid >> 5;
    const int wm     = (wid >> 2) * 64;   // wave M offset: 0 / 64
    const int wn     = (wid & 3) * 64;    // wave N offset: 0/64/128/192
    const int hl     = lane >> 4;         // half-wave select
    const int r      = lane & 15;         // row/col within 16

    const __hip_bfloat16* Ab = A  + (size_t)m_base * K_DIM;
    const __hip_bfloat16* Bb = Bm + (size_t)n_base * K_DIM;
    const unsigned ldsA_addr = (unsigned)(uintptr_t)(&ldsA[0]);
    const unsigned ldsB_addr = (unsigned)(uintptr_t)(&ldsB[0]);

    v8f acc[4][4];
#pragma unroll
    for (int i = 0; i < 4; ++i)
#pragma unroll
        for (int j = 0; j < 4; ++j)
            acc[i][j] = (v8f){0.f, 0.f, 0.f, 0.f, 0.f, 0.f, 0.f, 0.f};

    // ---- prologue: TDM-load tile 0 into buffer 0 ---------------------------
    if (wid == 0) {
        tdm_load_tile(Ab, ldsA_addr, 128, M_DIM);
        tdm_load_tile(Bb, ldsB_addr, 256, N_DIM);
        __builtin_amdgcn_s_wait_tensorcnt(0);
    }
    __syncthreads();

    for (int k0 = 0; k0 < K_DIM; k0 += K_STEP) {
        const int buf = (k0 >> 6) & 1;

        // kick off next tile's DMA into the other buffer (wave 0 only; the
        // barrier at the previous loop tail guarantees that buffer is free)
        if ((k0 + K_STEP < K_DIM) && wid == 0) {
            tdm_load_tile(Ab + k0 + K_STEP,
                          ldsA_addr + (unsigned)((buf ^ 1) * A_TILE_ELEMS * 2),
                          128, M_DIM);
            tdm_load_tile(Bb + k0 + K_STEP,
                          ldsB_addr + (unsigned)((buf ^ 1) * B_TILE_ELEMS * 2),
                          256, N_DIM);
        }

        const __hip_bfloat16* la = ldsA + buf * A_TILE_ELEMS;
        const __hip_bfloat16* lb = ldsB + buf * B_TILE_ELEMS;

        // ---- consume the 64-wide K slab as two 32-wide WMMA chunks --------
#pragma unroll
        for (int kc = 0; kc < K_STEP; kc += 32) {
            // B 32x16 bf16 frags: lanes 0-15 -> N=r with K kc..kc+15,
            // lanes 16-31 -> K kc+16..kc+31 (contiguous 32 B per lane).
            v16bf bfr[4];
#pragma unroll
            for (int j = 0; j < 4; ++j) {
                const int row = wn + j * 16 + r;
                const uint4* p = (const uint4*)(lb + row * LDS_STRIDE + kc + hl * 16);
                union { uint4 u[2]; v16bf v; } t;
                t.u[0] = p[0];
                t.u[1] = p[1];
                bfr[j] = t.v;
            }
            // A 16x32 bf16 frags: row M=r; low lanes hold
            // K kc+0-15, high lanes K kc+16-31 (matches B).
#pragma unroll
            for (int i = 0; i < 4; ++i) {
                const int row = wm + i * 16 + r;
                const uint4* p = (const uint4*)(la + row * LDS_STRIDE + kc + hl * 16);
                union { uint4 u[2]; v16bf v; } t;
                t.u[0] = p[0];      // +0 B
                t.u[1] = p[1];      // +16 B
                const v16bf afr = t.v;
#pragma unroll
                for (int j = 0; j < 4; ++j)
                    acc[i][j] = __builtin_amdgcn_wmma_f32_16x16x32_bf16(
                        /*neg_a=*/false, afr, /*neg_b=*/false, bfr[j],
                        /*c_mod=*/(short)0, acc[i][j],
                        /*reuse_a=*/false, /*reuse_b=*/false);
            }
        }

        // next buffer must be resident before anyone proceeds
        if (wid == 0) __builtin_amdgcn_s_wait_tensorcnt(0);
        __syncthreads();
    }

    // ---- epilogue: C/D layout -> row-major store, + bias --------------------
    // lane < 16: VGPR v -> M = v,   N = lane ; lane >= 16: M = v+8, N = lane-16
#pragma unroll
    for (int i = 0; i < 4; ++i) {
#pragma unroll
        for (int j = 0; j < 4; ++j) {
            const int col = n_base + wn + j * 16 + r;
            const float bv = bias[col];
#pragma unroll
            for (int v = 0; v < 8; ++v) {
                const int rowm = m_base + wm + i * 16 + hl * 8 + v;
                C[(size_t)rowm * N_DIM + col] = acc[i][j][v] + bv;
            }
        }
    }
}

// ---------------------------------------------------------------------------
// Host-side launch
// ---------------------------------------------------------------------------
extern "C" void kernel_launch(void* const* d_in, const int* in_sizes, int n_in,
                              void* d_out, int out_size, void* d_ws, size_t ws_size,
                              hipStream_t stream) {
    (void)in_sizes; (void)n_in; (void)out_size; (void)ws_size;

    const float* x    = (const float*)d_in[0];   // [B,S,H]
    const float* W    = (const float*)d_in[1];   // [O,H]
    const float* bias = (const float*)d_in[2];   // [O]
    float* out        = (float*)d_out;           // [B,S,O]

    // workspace: s bf16 [M][K] (64 MiB), W bf16 [N][K] (8 MiB), E f32 (1 MiB)
    __hip_bfloat16* sbf = (__hip_bfloat16*)d_ws;
    __hip_bfloat16* wbf = sbf + (size_t)M_DIM * K_DIM;
    float*          Ebuf = (float*)(wbf + (size_t)N_DIM * K_DIM);

    const int scan_threads = B_DIM * CHUNKS * H_DIM;           // 262144
    ema_partial_kernel<<<scan_threads / 256, 256, 0, stream>>>(x, Ebuf);
    ema_apply_kernel<<<scan_threads / 256, 256, 0, stream>>>(x, Ebuf, sbf);
    conv_w_kernel<<<(N_DIM * K_DIM) / 256, 256, 0, stream>>>(W, wbf, N_DIM * K_DIM);

    dim3 grid(N_DIM / 256, M_DIM / 128);   // (8, 128) = 1024 blocks
    wmma_gemm_kernel<<<grid, 256, 0, stream>>>(sbf, wbf, bias, out);
}
